// HopfieldModel_32469952758521
// MI455X (gfx1250) — matrix operand order repaired and checked by reading, not compiled, not run on hardware
//
#include <hip/hip_runtime.h>
#include <hip/hip_bf16.h>
#include <stdint.h>

// ---------------------------------------------------------------------------
// Hopfield retrieval network, fused for MI455X (gfx1250, wave32, WMMA + TDM).
//   z = x@W_enc + b_enc                       (tiny fp32 GEMV kernel)
//   keys -> bf16 once (33.5MB, lives in the 192MB L2)
//   4x fixed-point steps of flash-attention:  q <- softmax(2 q K^T) K
//     - v_wmma_f32_16x16x32_bf16 for both GEMMs, fp32 accum
//     - key tiles staged to LDS by the Tensor Data Mover (tensor_load_to_lds,
//       TENSORcnt), with D# padding giving a bank-conflict-free row stride
//     - online softmax, K split into 8 partitions, LSE-merged by combine
//   logits = log_softmax(relu(concat q) @ W_lin + b_lin)
// ---------------------------------------------------------------------------

#define N_NETS     4
#define BATCH      256
#define IN_DIM     784
#define EDIM       128
#define N_EMB      32768
#define STEPS      4
#define NKP        8            // K partitions -> 4*4*8 = 128 workgroups
#define KPP        (N_EMB/NKP)  // 4096 keys per partition
#define TILE_K     32           // keys per inner tile (one WMMA contraction)
#define KROW       136          // padded LDS row stride (halfs): 128 + 8 (TDM pad)

typedef __attribute__((ext_vector_type(8)))  float        v8f;
typedef __attribute__((ext_vector_type(16))) __bf16       v16bf;
typedef __attribute__((ext_vector_type(4)))  unsigned int u32x4;
typedef __attribute__((ext_vector_type(8)))  int          i32x8;
typedef __attribute__((ext_vector_type(4)))  int          i32x4;

union BfVec {                 // 16 bf16 = 32 bytes = 8 VGPRs (WMMA A/B operand)
    u32x4 u[2];
    v16bf v;
};

static __device__ __forceinline__ unsigned short f2bf(float f) {
    unsigned int u = __float_as_uint(f);
    u += 0x7FFFu + ((u >> 16) & 1u);   // round-to-nearest-even
    return (unsigned short)(u >> 16);
}

static __device__ __forceinline__ void wait_tensorcnt0() {
#if __has_builtin(__builtin_amdgcn_s_wait_tensorcnt)
    __builtin_amdgcn_s_wait_tensorcnt(0);
#else
    asm volatile("s_wait_tensorcnt 0x0" ::: "memory");
#endif
}

// -------------------------------------------------------------- encoder ----
__global__ __launch_bounds__(128)
void encode_init_kernel(const float* __restrict__ x,
                        const float* __restrict__ W_enc,
                        const float* __restrict__ b_enc,
                        unsigned short* __restrict__ qscaled) { // [4][256][128] bf16 of 2*z
    const int b = blockIdx.x;
    const int e = threadIdx.x;
    float acc = b_enc[e];
    const float* xr = x + (size_t)b * IN_DIM;
    for (int i = 0; i < IN_DIM; ++i)
        acc = fmaf(xr[i], W_enc[(size_t)i * EDIM + e], acc);
    unsigned short h = f2bf(2.0f * acc);   // fold beta=2.0 into the query
    #pragma unroll
    for (int n = 0; n < N_NETS; ++n)
        qscaled[((size_t)n * BATCH + b) * EDIM + e] = h;
}

// --------------------------------------------------------- keys -> bf16 ----
__global__ __launch_bounds__(256)
void convert_keys_kernel(const float* __restrict__ emb,      // [4*32768][129]
                         unsigned short* __restrict__ keys) { // [4*32768][128] bf16
    const size_t total  = (size_t)N_NETS * N_EMB * EDIM;
    const size_t stride = (size_t)gridDim.x * blockDim.x;
    for (size_t i = (size_t)blockIdx.x * blockDim.x + threadIdx.x; i < total; i += stride) {
        size_t row = i >> 7;          // /128
        size_t e   = i & 127;
        keys[i] = f2bf(emb[row * (EDIM + 1) + e]);
    }
}

// --------------------------------------------- flash-attention partial -----
// grid = (NKP, 4 query-blocks of 64, N_NETS); block = 128 (4 waves of 32).
// Each wave: 16 query rows, streams its 4096-key partition in 32-key tiles.
__global__ __launch_bounds__(128)
void hopfield_attn_partial(const unsigned short* __restrict__ qscaled, // bf16 2*q
                           const unsigned short* __restrict__ keys,    // bf16
                           float* __restrict__ mpart,   // [4][NKP][256]
                           float* __restrict__ lpart,   // [4][NKP][256]
                           float* __restrict__ accpart) // [4][NKP][256][128]
{
    const int kp   = blockIdx.x;
    const int qb   = blockIdx.y;
    const int n    = blockIdx.z;
    const int tid  = threadIdx.x;
    const int wv   = tid >> 5;
    const int lane = tid & 31;
    const int l16  = lane & 15;
    const int lhi  = lane >> 4;      // which 16-lane half

    __shared__ __align__(16) unsigned short skrow[TILE_K * KROW]; // key tile [k][e], TDM-padded
    __shared__ __align__(16) unsigned short skeyT[EDIM * 40];     // keys^T tile [e][k], padded
    __shared__ __align__(16) unsigned short sp[4 * 16 * 40];      // per-wave P tile [16][32], padded

    // ---- preload A operand (queries, pre-scaled by beta), ISA A-layout:
    //      row = lane%16, chunks of 8 halfs at K = lhi*8 and 16+lhi*8.
    const int qrow0 = qb * 64 + wv * 16 + l16;
    const unsigned short* qptr = qscaled + ((size_t)n * BATCH + qrow0) * EDIM;
    BfVec a[4];
    #pragma unroll
    for (int kk = 0; kk < 4; ++kk) {
        const unsigned short* pa = qptr + kk * 32 + lhi * 8;
        a[kk].u[0] = *(const u32x4*)(pa);
        a[kk].u[1] = *(const u32x4*)(pa + 16);
    }

    float mrow[8], lrow[8];
    v8f   O[8];
    #pragma unroll
    for (int r = 0; r < 8; ++r) { mrow[r] = -__builtin_inff(); lrow[r] = 0.0f; }
    #pragma unroll
    for (int es = 0; es < 8; ++es) {
        v8f z = {0.f,0.f,0.f,0.f,0.f,0.f,0.f,0.f};
        O[es] = z;
    }

    const unsigned short* kbase = keys + ((size_t)n * N_EMB + (size_t)kp * KPP) * EDIM;
    unsigned short* spw = sp + wv * 16 * 40;

    for (int t = 0; t < KPP / TILE_K; ++t) {
        const unsigned short* ktile = kbase + (size_t)t * TILE_K * EDIM;

        __syncthreads();                         // previous tile fully consumed

        // ---- stage key tile [32][128] bf16 -> LDS row-major (stride KROW)
#if __has_builtin(__builtin_amdgcn_tensor_load_to_lds)
        if (wv == 0) {
            // Tensor DMA descriptor (D#), 2-D tensor, pad 4 DWORDs per 64 DWORDs
            // so each 256B key row lands at a 272B (136-half) LDS stride.
            unsigned long long ga = (unsigned long long)(size_t)ktile;
            u32x4 g0;
            g0[0] = 1u;                                   // count=1 (valid D#)
            g0[1] = (unsigned int)(size_t)skrow;          // lds_addr (bytes)
            g0[2] = (unsigned int)ga;                     // global_addr[31:0]
            g0[3] = (unsigned int)(ga >> 32) | (2u << 30);// global_addr[56:32] | type=2
            i32x8 g1;
            g1[0] = (1 << 16)      // data_size = 1 -> 2 bytes
                  | (1 << 20)      // pad_enable
                  | (5 << 22)      // pad_interval: 64 DWORDs (= one 256B row)
                  | (3 << 25);     // pad_amount:   4 DWORDs (= 16B = 8 halfs)
            g1[1] = (int)(128u << 16);   // tensor_dim0 = 128 (elements)
            g1[2] = (int)(32u  << 16);   // tensor_dim1 = 32
            g1[3] = (int)(128u << 16);   // tile_dim0   = 128
            g1[4] = 32;                  // tile_dim1   = 32
            g1[5] = 128;                 // tensor_dim0_stride = 128 elements
            g1[6] = 0;
            g1[7] = 0;
            i32x4 gz4 = {0, 0, 0, 0};                  // groups 2/3 unused (2-D)
            i32x8 gz8 = {0, 0, 0, 0, 0, 0, 0, 0};      // extra group (clang-23 form)
            __builtin_amdgcn_tensor_load_to_lds(g0, g1, gz4, gz4, gz8, 0);
            wait_tensorcnt0();
        }
#else
        // Fallback: cooperative vectorized copy global -> LDS.
        #pragma unroll
        for (int i = 0; i < 4; ++i) {
            int idx = tid + i * 128;            // 0..511 -> 32 keys x 16 e-chunks
            int k   = idx >> 4;
            int e0  = (idx & 15) << 3;
            *(u32x4*)(skrow + k * KROW + e0) = *(const u32x4*)(ktile + k * EDIM + e0);
        }
#endif
        __syncthreads();

        // ---- build transposed copy (PV B operand) from the LDS row tile
        #pragma unroll
        for (int i = 0; i < 4; ++i) {
            int idx = tid + i * 128;            // 0..511 -> 32 keys x 16 e-chunks
            int k   = idx >> 4;
            int e0  = (idx & 15) << 3;
            u32x4 v = *(const u32x4*)(skrow + k * KROW + e0);
            const unsigned short* hp = (const unsigned short*)&v;
            #pragma unroll
            for (int j = 0; j < 8; ++j)
                skeyT[(e0 + j) * 40 + k] = hp[j];
        }
        __syncthreads();

        if (t + 1 < KPP / TILE_K)               // global_prefetch_b8 next tile into L2
            __builtin_prefetch(ktile + TILE_K * EDIM + tid * 32, 0, 1);

        // ---- S = (2q) K^T for this 32-key tile; two 16-col C halves.
        //      B-layout: col = lane%16, chunks of 8 halfs at K = lhi*8 and 16+lhi*8.
        v8f c0 = {0.f,0.f,0.f,0.f,0.f,0.f,0.f,0.f};
        v8f c1 = c0;
        #pragma unroll
        for (int kk = 0; kk < 4; ++kk) {
            BfVec b0, b1;
            const unsigned short* kr0 = skrow + (l16)      * KROW + kk * 32 + lhi * 8;
            const unsigned short* kr1 = skrow + (16 + l16) * KROW + kk * 32 + lhi * 8;
            b0.u[0] = *(const u32x4*)(kr0); b0.u[1] = *(const u32x4*)(kr0 + 16);
            b1.u[0] = *(const u32x4*)(kr1); b1.u[1] = *(const u32x4*)(kr1 + 16);
            c0 = __builtin_amdgcn_wmma_f32_16x16x32_bf16(false, a[kk].v, false, b0.v,
                                                         (short)0, c0, false, false);
            c1 = __builtin_amdgcn_wmma_f32_16x16x32_bf16(false, a[kk].v, false, b1.v,
                                                         (short)0, c1, false, false);
        }

        // ---- online softmax. C layout: elem r = row r+8*lhi, col l16 (+16 for c1).
        float tm[8];
        #pragma unroll
        for (int r = 0; r < 8; ++r) tm[r] = fmaxf(c0[r], c1[r]);
        #pragma unroll
        for (int msk = 1; msk < 16; msk <<= 1) {
            #pragma unroll
            for (int r = 0; r < 8; ++r)
                tm[r] = fmaxf(tm[r], __shfl_xor(tm[r], msk, 32));
        }
        float sc[8];
        #pragma unroll
        for (int r = 0; r < 8; ++r) {
            float nm = fmaxf(mrow[r], tm[r]);
            sc[r]    = __expf(mrow[r] - nm);
            mrow[r]  = nm;
            lrow[r] *= sc[r];
        }
        #pragma unroll
        for (int es = 0; es < 8; ++es) {
            #pragma unroll
            for (int r = 0; r < 8; ++r) O[es][r] *= sc[r];
        }

        float rs[8];
        #pragma unroll
        for (int r = 0; r < 8; ++r) {
            float p0 = __expf(c0[r] - mrow[r]);
            float p1 = __expf(c1[r] - mrow[r]);
            rs[r] = p0 + p1;
            int row = r + lhi * 8;
            spw[row * 40 + l16]      = f2bf(p0);
            spw[row * 40 + 16 + l16] = f2bf(p1);
        }
        #pragma unroll
        for (int msk = 1; msk < 16; msk <<= 1) {
            #pragma unroll
            for (int r = 0; r < 8; ++r)
                rs[r] += __shfl_xor(rs[r], msk, 32);
        }
        #pragma unroll
        for (int r = 0; r < 8; ++r) lrow[r] += rs[r];

        // ---- O += P * K_tile (contraction over 32 keys), 8 e-slices.
        BfVec pA;
        const unsigned short* pr = spw + l16 * 40 + lhi * 8;   // A-layout reload of P
        pA.u[0] = *(const u32x4*)(pr);
        pA.u[1] = *(const u32x4*)(pr + 16);
        #pragma unroll
        for (int es = 0; es < 8; ++es) {
            BfVec bT;
            const unsigned short* tr = skeyT + (es * 16 + l16) * 40 + lhi * 8;
            bT.u[0] = *(const u32x4*)(tr);
            bT.u[1] = *(const u32x4*)(tr + 16);
            O[es] = __builtin_amdgcn_wmma_f32_16x16x32_bf16(false, pA.v, false, bT.v,
                                                            (short)0, O[es], false, false);
        }
    }

    // ---- write partials for the LSE merge
    const int base = (n * NKP + kp) * BATCH;
    if (l16 == 0) {
        #pragma unroll
        for (int r = 0; r < 8; ++r) {
            int row = qb * 64 + wv * 16 + r + lhi * 8;
            mpart[base + row] = mrow[r];
            lpart[base + row] = lrow[r];
        }
    }
    #pragma unroll
    for (int es = 0; es < 8; ++es) {
        #pragma unroll
        for (int r = 0; r < 8; ++r) {
            int row = qb * 64 + wv * 16 + r + lhi * 8;
            accpart[((size_t)(base + row)) * EDIM + es * 16 + l16] = O[es][r];
        }
    }
}

// ---------------------------------------------------- LSE merge / update ---
__global__ __launch_bounds__(128)
void hopfield_combine(const float* __restrict__ mpart,
                      const float* __restrict__ lpart,
                      const float* __restrict__ accpart,
                      float* __restrict__ qf32,            // [4][256][128]
                      unsigned short* __restrict__ qscaled) // bf16 of 2*q
{
    const int n   = blockIdx.x >> 8;
    const int row = blockIdx.x & 255;
    const int e   = threadIdx.x;
    float M = -__builtin_inff();
    #pragma unroll
    for (int p = 0; p < NKP; ++p)
        M = fmaxf(M, mpart[(n * NKP + p) * BATCH + row]);
    float L = 0.0f, A = 0.0f;
    #pragma unroll
    for (int p = 0; p < NKP; ++p) {
        float w = __expf(mpart[(n * NKP + p) * BATCH + row] - M);
        L += lpart[(n * NKP + p) * BATCH + row] * w;
        A += accpart[((size_t)((n * NKP + p) * BATCH + row)) * EDIM + e] * w;
    }
    float q = A / L;
    size_t o = ((size_t)n * BATCH + row) * EDIM + e;
    qf32[o]    = q;
    qscaled[o] = f2bf(2.0f * q);
}

// ----------------------------------------------------------- classifier ----
__global__ __launch_bounds__(256)
void classifier_kernel(const float* __restrict__ qf32,
                       const float* __restrict__ W_lin,   // [512][10]
                       const float* __restrict__ b_lin,
                       float* __restrict__ out) {          // [256][10]
    const int b = threadIdx.x;
    float lg[10];
    #pragma unroll
    for (int c = 0; c < 10; ++c) lg[c] = b_lin[c];
    for (int j = 0; j < N_NETS * EDIM; ++j) {
        int nn = j >> 7, e = j & 127;
        float v = qf32[((size_t)nn * BATCH + b) * EDIM + e];
        v = fmaxf(v, 0.0f);
        #pragma unroll
        for (int c = 0; c < 10; ++c) lg[c] = fmaf(v, W_lin[j * 10 + c], lg[c]);
    }
    float M = lg[0];
    #pragma unroll
    for (int c = 1; c < 10; ++c) M = fmaxf(M, lg[c]);
    float s = 0.0f;
    #pragma unroll
    for (int c = 0; c < 10; ++c) s += __expf(lg[c] - M);
    float lse = M + __logf(s);
    #pragma unroll
    for (int c = 0; c < 10; ++c) out[b * 10 + c] = lg[c] - lse;
}

// ---------------------------------------------------------------------------
extern "C" void kernel_launch(void* const* d_in, const int* in_sizes, int n_in,
                              void* d_out, int out_size, void* d_ws, size_t ws_size,
                              hipStream_t stream) {
    const float* x     = (const float*)d_in[0];
    const float* emb   = (const float*)d_in[1];
    const float* W_enc = (const float*)d_in[2];
    const float* b_enc = (const float*)d_in[3];
    const float* W_lin = (const float*)d_in[4];
    const float* b_lin = (const float*)d_in[5];
    float* out = (float*)d_out;

    char* ws = (char*)d_ws;
    unsigned short* keysbf  = (unsigned short*)(ws);                 // 33,554,432 B
    unsigned short* qscaled = (unsigned short*)(ws + 33554432);      //    262,144 B
    float*          qf32    = (float*)         (ws + 33816576);      //    524,288 B
    float*          mpart   = (float*)         (ws + 34340864);      //     32,768 B
    float*          lpart   = (float*)         (ws + 34373632);      //     32,768 B
    float*          accpart = (float*)         (ws + 34406400);      //  4,194,304 B
    // total ws usage: 38,600,704 B

    encode_init_kernel<<<dim3(BATCH), 128, 0, stream>>>(x, W_enc, b_enc, qscaled);
    convert_keys_kernel<<<dim3(4096), 256, 0, stream>>>(emb, keysbf);

    for (int s = 0; s < STEPS; ++s) {
        hopfield_attn_partial<<<dim3(NKP, 4, N_NETS), 128, 0, stream>>>(
            qscaled, keysbf, mpart, lpart, accpart);
        hopfield_combine<<<dim3(N_NETS * BATCH), 128, 0, stream>>>(
            mpart, lpart, accpart, qf32, qscaled);
    }

    classifier_kernel<<<dim3(1), 256, 0, stream>>>(qf32, W_lin, b_lin, out);
}
